// GNN_34986803593596
// MI455X (gfx1250) — matrix operand order repaired and checked by reading, not compiled, not run on hardware
//
#include <hip/hip_runtime.h>
#include <hip/hip_bf16.h>

// ---------------------------------------------------------------------------
// 3-layer GCN on gfx1250:
//   deg/dinv + edge-meta precompute -> per layer:
//     [bf16 WMMA GEMM, B staged in LDS via async DMA + double-buffered reads,
//      epilogue writes H and Out=bias+dinv^2*H]
//     [edge scatter: gather H[src], scale, 4x f32 atomics into Out[dst]]
//   ReLU of intermediate layers folded into next GEMM's A-load; final ReLU pass.
// ---------------------------------------------------------------------------

typedef __bf16 v16bf __attribute__((ext_vector_type(16)));
typedef float  v8f   __attribute__((ext_vector_type(8)));
typedef int    v4i   __attribute__((ext_vector_type(4)));

#define AS_GLOBAL __attribute__((address_space(1)))
#define AS_SHARED __attribute__((address_space(3)))

union BF16Frag {
    v16bf v;
    unsigned int u[8];
};

// ---------------------------------------------------------------------------
// Degree / normalization kernels
// ---------------------------------------------------------------------------
__global__ void deg_init_kernel(float* __restrict__ deg, int N) {
    int i = blockIdx.x * blockDim.x + threadIdx.x;
    if (i < N) deg[i] = 1.0f;  // self loop contributes 1
}

__global__ void deg_accum_kernel(const long long* __restrict__ dsts,
                                 float* __restrict__ deg, long long E) {
    long long e = (long long)blockIdx.x * blockDim.x + threadIdx.x;
    if (e < E) unsafeAtomicAdd(&deg[(int)dsts[e]], 1.0f);
}

__global__ void deg_rsqrt_kernel(float* __restrict__ deg, int N) {
    int i = blockIdx.x * blockDim.x + threadIdx.x;
    if (i < N) deg[i] = rsqrtf(deg[i]);  // deg >= 1 always
}

// Pack per-edge metadata once: int32 endpoints + combined norm factor.
__global__ void edge_prep_kernel(const long long* __restrict__ srcs,
                                 const long long* __restrict__ dsts,
                                 const float* __restrict__ dinv,
                                 int* __restrict__ s32, int* __restrict__ d32,
                                 float* __restrict__ nrm, long long E) {
    long long e = (long long)blockIdx.x * blockDim.x + threadIdx.x;
    if (e >= E) return;
    int s = (int)srcs[e];
    int d = (int)dsts[e];
    s32[e] = s;
    d32[e] = d;
    nrm[e] = dinv[s] * dinv[d];
}

// ---------------------------------------------------------------------------
// Repack W[3][128][128] (f32 row-major, W[k][n]) into WMMA B-fragment layout,
// bf16. One wave per (layer, ktile, ntile) fragment: lane holds col=lane%16,
// K = ktile*32 + (lane<16 ? 0..15 : 16..31), 2 bf16 per dword. Fragment is
// 32 lanes * 32B contiguous so the GEMM stages it with one DMA burst.
// ---------------------------------------------------------------------------
__global__ __launch_bounds__(256) void repack_w_kernel(
    const float* __restrict__ Ws, unsigned short* __restrict__ Wb) {
    int wave = (blockIdx.x * blockDim.x + threadIdx.x) >> 5;  // 0..95
    int lane = threadIdx.x & 31;
    int layer = wave >> 5;       // 0..2
    int rem   = wave & 31;
    int ktile = rem >> 3;        // 0..3
    int ntile = rem & 7;         // 0..7
    const float* W = Ws + (size_t)layer * 128 * 128;
    int col = ntile * 16 + (lane & 15);
    int k0  = ktile * 32 + (lane >> 4) * 16;
    BF16Frag frag;
#pragma unroll
    for (int j = 0; j < 16; ++j)
        frag.v[j] = (__bf16)W[(size_t)(k0 + j) * 128 + col];
    unsigned int* dst = (unsigned int*)Wb +
        ((size_t)(layer * 32 + ktile * 8 + ntile) * 32 + lane) * 8;
#pragma unroll
    for (int v = 0; v < 8; ++v) dst[v] = frag.u[v];
}

// ---------------------------------------------------------------------------
// GEMM + fused epilogue:
//   H[r][c]   = act(X) @ W          (bf16 WMMA, f32 accum)
//   Out[r][c] = bias[c] + dinv[r]^2 * H[r][c]   (bias + self-loop term)
// Block = 256 threads = 8 waves; wave handles 16 rows x 128 cols.
// B fragments (32KB/layer) staged once per block into LDS via async DMA,
// then double-buffered through ds_load_b128 across the nt loop.
// ---------------------------------------------------------------------------
template <bool GUARD>
__global__ __launch_bounds__(256) void gcn_gemm_kernel(
    const float* __restrict__ X, const unsigned int* __restrict__ Wb32,
    const float* __restrict__ bias, const float* __restrict__ dinv,
    float* __restrict__ H, float* __restrict__ Out, int N, int rowbase0,
    int relu_in) {
    __shared__ unsigned int smemW[32 * 32 * 8];  // 32 frags * 1KB = 32KB

    const int tid = threadIdx.x;
    // ---- stage this layer's B fragments into LDS (async DMA if available) --
#pragma unroll
    for (int i = 0; i < 8; ++i) {
        int dw = i * 1024 + tid * 4;  // 4 dwords (b128) per thread per iter
#if __has_builtin(__builtin_amdgcn_global_load_async_to_lds_b128)
        __builtin_amdgcn_global_load_async_to_lds_b128(
            (AS_GLOBAL v4i*)(Wb32 + dw),
            (AS_SHARED v4i*)(smemW + dw),
            /*offset=*/0, /*cpol=*/0);
#else
        *(uint4*)(smemW + dw) = *(const uint4*)(Wb32 + dw);
#endif
    }
#if __has_builtin(__builtin_amdgcn_s_wait_asynccnt)
    __builtin_amdgcn_s_wait_asynccnt(0);
#else
    asm volatile("s_wait_asynccnt 0x0" ::: "memory");
#endif
    __syncthreads();

    const int lane = tid & 31;
    const bool hiHalf = lane >= 16;
    const int rowbase = rowbase0 + blockIdx.x * 128 + (tid >> 5) * 16;
    int row = rowbase + (lane & 15);
    if (GUARD && row >= N) row = N - 1;  // clamp: EXEC stays all-ones for WMMA

    // ---- A fragments: native f32->bf16 converts (v_cvt_pk_bf16_f32) -------
    BF16Frag a[4];
#pragma unroll
    for (int kt = 0; kt < 4; ++kt) {
        const float* rp = X + (size_t)row * 128 + kt * 32 + (hiHalf ? 16 : 0);
        float4 t0 = *(const float4*)(rp);
        float4 t1 = *(const float4*)(rp + 4);
        float4 t2 = *(const float4*)(rp + 8);
        float4 t3 = *(const float4*)(rp + 12);
        float f[16] = {t0.x, t0.y, t0.z, t0.w, t1.x, t1.y, t1.z, t1.w,
                       t2.x, t2.y, t2.z, t2.w, t3.x, t3.y, t3.z, t3.w};
        if (relu_in) {
#pragma unroll
            for (int j = 0; j < 16; ++j) f[j] = fmaxf(f[j], 0.0f);
        }
#pragma unroll
        for (int j = 0; j < 16; ++j) a[kt].v[j] = (__bf16)f[j];
    }

    // ---- per-lane self-loop scales for the 8 rows this lane stores --------
    float dv[8];
    if (!GUARD) {
        // rows are contiguous: two broadcast float4 loads, no per-row chains
        const float* dbase = dinv + rowbase + (hiHalf ? 8 : 0);
        float4 d0 = *(const float4*)(dbase);
        float4 d1 = *(const float4*)(dbase + 4);
        dv[0] = d0.x * d0.x; dv[1] = d0.y * d0.y;
        dv[2] = d0.z * d0.z; dv[3] = d0.w * d0.w;
        dv[4] = d1.x * d1.x; dv[5] = d1.y * d1.y;
        dv[6] = d1.z * d1.z; dv[7] = d1.w * d1.w;
    } else {
#pragma unroll
        for (int i = 0; i < 8; ++i) {
            int r = rowbase + (hiHalf ? i + 8 : i);
            if (r >= N) r = N - 1;
            float di = dinv[r];
            dv[i] = di * di;
        }
    }

    // ---- preload the 8 bias values this lane needs (one clause) -----------
    float bcv[8];
#pragma unroll
    for (int nt = 0; nt < 8; ++nt) bcv[nt] = bias[nt * 16 + (lane & 15)];

    // ---- main loop: double-buffered B fragments from LDS ------------------
    const unsigned int* sw = smemW + lane * 8;
    BF16Frag bb[2][4];
#pragma unroll
    for (int kt = 0; kt < 4; ++kt) {
        const unsigned int* bp = sw + (kt * 8) * 32 * 8;  // nt = 0
#pragma unroll
        for (int v = 0; v < 8; ++v) bb[0][kt].u[v] = bp[v];
    }

    const size_t rbase = (size_t)(rowbase + (hiHalf ? 8 : 0)) * 128;
#pragma unroll
    for (int nt = 0; nt < 8; ++nt) {
        // prefetch next nt's 4 fragments while this nt's WMMAs run
        if (nt < 7) {
#pragma unroll
            for (int kt = 0; kt < 4; ++kt) {
                const unsigned int* bp = sw + (kt * 8 + nt + 1) * 32 * 8;
#pragma unroll
                for (int v = 0; v < 8; ++v)
                    bb[(nt + 1) & 1][kt].u[v] = bp[v];
            }
        }
        v8f c = {};
#pragma unroll
        for (int kt = 0; kt < 4; ++kt)
            c = __builtin_amdgcn_wmma_f32_16x16x32_bf16(
                    false, a[kt].v, false, bb[nt & 1][kt].v, (short)0, c,
                    false, false);

        const int col = nt * 16 + (lane & 15);
        const float bc = bcv[nt];
        if (!GUARD) {
            float* hp = H + rbase + col;
            float* op = Out + rbase + col;
#pragma unroll
            for (int i = 0; i < 8; ++i) {  // immediate-offset strided stores
                float h = c[i];
                hp[i * 128] = h;
                op[i * 128] = bc + dv[i] * h;
            }
        } else {
#pragma unroll
            for (int i = 0; i < 8; ++i) {
                int r = rowbase + (hiHalf ? i + 8 : i);
                if (r < N) {
                    float h = c[i];
                    H[(size_t)r * 128 + col]   = h;
                    Out[(size_t)r * 128 + col] = bc + dv[i] * h;
                }
            }
        }
    }
}

// ---------------------------------------------------------------------------
// Edge scatter: one wave per edge; float4 gather of H[src] (L2 resident),
// scale by precomputed norm, 4x native f32 atomic adds into Out[dst].
// ---------------------------------------------------------------------------
__global__ __launch_bounds__(256) void scatter_kernel(
    const int* __restrict__ s32, const int* __restrict__ d32,
    const float* __restrict__ nrm, const float* __restrict__ H,
    float* __restrict__ Out, long long E) {
    long long wave = ((long long)blockIdx.x * blockDim.x + threadIdx.x) >> 5;
    int lane = threadIdx.x & 31;
    if (wave >= E) return;
    int s = s32[wave];
    int d = d32[wave];
    float norm = nrm[wave];
    const float4 hv = *(const float4*)(H + (size_t)s * 128 + lane * 4);
    float* o = Out + (size_t)d * 128 + lane * 4;
    unsafeAtomicAdd(o + 0, hv.x * norm);
    unsafeAtomicAdd(o + 1, hv.y * norm);
    unsafeAtomicAdd(o + 2, hv.z * norm);
    unsafeAtomicAdd(o + 3, hv.w * norm);
}

__global__ __launch_bounds__(256) void relu_kernel(float* __restrict__ X,
                                                   long long total) {
    long long idx = (long long)blockIdx.x * blockDim.x + threadIdx.x;
    if (idx < total) X[idx] = fmaxf(X[idx], 0.0f);
}

// ---------------------------------------------------------------------------
extern "C" void kernel_launch(void* const* d_in, const int* in_sizes, int n_in,
                              void* d_out, int out_size, void* d_ws,
                              size_t ws_size, hipStream_t stream) {
    const float* x      = (const float*)d_in[0];
    const long long* ei = (const long long*)d_in[1];
    const float* Ws     = (const float*)d_in[2];
    const float* bs     = (const float*)d_in[3];

    const int N = in_sizes[0] / 128;
    const long long E = (long long)in_sizes[1] / 2;
    const long long* srcs = ei;
    const long long* dsts = ei + E;

    // workspace layout
    char* ws = (char*)d_ws;
    size_t off = 0;
    auto take = [&](size_t bytes) -> void* {
        void* p = ws + off;
        off += (bytes + 255) & ~(size_t)255;
        return p;
    };
    float* dinv        = (float*)take((size_t)N * 4);
    float* H           = (float*)take((size_t)N * 128 * 4);
    float* xbuf        = (float*)take((size_t)N * 128 * 4);
    unsigned short* Wb = (unsigned short*)take((size_t)3 * 128 * 128 * 2);
    int* s32           = (int*)take((size_t)E * 4);
    int* d32           = (int*)take((size_t)E * 4);
    float* nrm         = (float*)take((size_t)E * 4);

    const dim3 blk(256);
    const long long total = (long long)N * 128;
    const unsigned node_grid = (unsigned)((N + 255) / 256);
    const unsigned edgeflat_grid = (unsigned)((E + 255) / 256);

    // normalization + edge-meta precompute
    deg_init_kernel<<<node_grid, blk, 0, stream>>>(dinv, N);
    deg_accum_kernel<<<edgeflat_grid, blk, 0, stream>>>(dsts, dinv, E);
    deg_rsqrt_kernel<<<node_grid, blk, 0, stream>>>(dinv, N);
    edge_prep_kernel<<<edgeflat_grid, blk, 0, stream>>>(srcs, dsts, dinv,
                                                        s32, d32, nrm, E);

    // repack all three layer weights into WMMA B-fragment layout (bf16)
    repack_w_kernel<<<12, blk, 0, stream>>>(Ws, Wb);

    const int full_blocks = N >> 7;            // guard-free 128-row tiles
    const int tail_rows   = N - (full_blocks << 7);
    const unsigned elem_grid = (unsigned)((total + 255) / 256);
    const unsigned edge_grid = (unsigned)((E + 7) / 8);

    const float* xin = x;
    for (int l = 0; l < 3; ++l) {
        float* out = (l == 2) ? (float*)d_out : xbuf;
        const unsigned int* wb32 =
            (const unsigned int*)(Wb + (size_t)l * 128 * 128);
        const float* bl = bs + (size_t)l * 128;
        if (full_blocks > 0)
            gcn_gemm_kernel<false><<<full_blocks, blk, 0, stream>>>(
                xin, wb32, bl, dinv, H, out, N, 0, l > 0 ? 1 : 0);
        if (tail_rows > 0)
            gcn_gemm_kernel<true><<<1, blk, 0, stream>>>(
                xin, wb32, bl, dinv, H, out, N, full_blocks << 7,
                l > 0 ? 1 : 0);
        scatter_kernel<<<edge_grid, blk, 0, stream>>>(
            s32, d32, nrm, H, out, E);
        xin = out;
    }
    // intermediate ReLUs were folded into the next GEMM's load; final one here
    relu_kernel<<<elem_grid, blk, 0, stream>>>((float*)d_out, total);
}
